// SkipEncoderModel_80934363726143
// MI455X (gfx1250) — matrix operand order repaired and checked by reading, not compiled, not run on hardware
//
#include <hip/hip_runtime.h>
#include <hip/hip_bf16.h>
#include <stdint.h>

// ---------------- model constants ----------------
#define BB   4
#define NS   96
#define TT   8
#define HH   256
#define LL   768
#define NN   97            // NS + 1
#define NLAY 6
#define H4   1024
#define M1   384           // B*NS (valid rows)
#define M1P  512           // pad256(M1)
#define MS   37636         // B*N*N
#define MSP  37888         // pad256(MS) = 148*256
#define MH   388           // B*N
#define MHP  512           // pad256(MH)

typedef __attribute__((ext_vector_type(16))) _Float16 v16h;
typedef __attribute__((ext_vector_type(8)))  _Float16 v8h;
typedef __attribute__((ext_vector_type(8)))  float    v8f;

__device__ __forceinline__ float sigf(float x) { return 1.0f / (1.0f + expf(-x)); }

__device__ __forceinline__ v16h cat8(v8h a, v8h b) {
  return __builtin_shufflevector(a, b, 0,1,2,3,4,5,6,7,8,9,10,11,12,13,14,15);
}

// ---- WMMA GEMM: C[M,Nc] f32 = A[M,256] f16 * Bt[Nc,256]^T f16 ----
// Block = 256 thr (8 waves). Each wave: 32 rows x 64 cols (8 accumulators).
// grid = (Nc/64, M/256). M multiple of 256, Nc multiple of 64, K = 256.
__global__ __launch_bounds__(256) void gemm_wmma(const _Float16* __restrict__ A,
                                                 const _Float16* __restrict__ Bt,
                                                 float* __restrict__ C, int Nc) {
  const int K = 256;
  int lane = threadIdx.x & 31;
  int wave = threadIdx.x >> 5;
  int r    = lane & 15;
  int hs   = lane >> 4;                       // half select
  size_t m0 = (size_t)blockIdx.y * 256 + (size_t)wave * 32;
  int    n0 = blockIdx.x * 64;
  // A fragment: lane holds row m0+r, K = {0..15} (hs=0) / {16..31} (hs=1)
  const _Float16* ap  = A  + (m0 + (size_t)r) * K + hs * 16;
  // B fragment: lane holds col n0+16j+r, K = {0..15} (hs=0) / {16..31} (hs=1)
  const _Float16* bp0 = Bt + ((size_t)(n0 + r)) * K + hs * 16;
  v8f acc00 = {}, acc01 = {}, acc02 = {}, acc03 = {};
  v8f acc10 = {}, acc11 = {}, acc12 = {}, acc13 = {};
#pragma unroll
  for (int kk = 0; kk < K; kk += 32) {
    v16h a0 = cat8(*(const v8h*)(ap + kk),            *(const v8h*)(ap + kk + 8));
    v16h a1 = cat8(*(const v8h*)(ap + 16 * K + kk),   *(const v8h*)(ap + 16 * K + kk + 8));
    v16h b0 = cat8(*(const v8h*)(bp0 + kk),           *(const v8h*)(bp0 + kk + 8));
    v16h b1 = cat8(*(const v8h*)(bp0 + 16 * K + kk),  *(const v8h*)(bp0 + 16 * K + kk + 8));
    v16h b2 = cat8(*(const v8h*)(bp0 + 32 * K + kk),  *(const v8h*)(bp0 + 32 * K + kk + 8));
    v16h b3 = cat8(*(const v8h*)(bp0 + 48 * K + kk),  *(const v8h*)(bp0 + 48 * K + kk + 8));
    acc00 = __builtin_amdgcn_wmma_f32_16x16x32_f16(false, a0, false, b0, (short)0, acc00, false, false);
    acc10 = __builtin_amdgcn_wmma_f32_16x16x32_f16(false, a1, false, b0, (short)0, acc10, false, false);
    acc01 = __builtin_amdgcn_wmma_f32_16x16x32_f16(false, a0, false, b1, (short)0, acc01, false, false);
    acc11 = __builtin_amdgcn_wmma_f32_16x16x32_f16(false, a1, false, b1, (short)0, acc11, false, false);
    acc02 = __builtin_amdgcn_wmma_f32_16x16x32_f16(false, a0, false, b2, (short)0, acc02, false, false);
    acc12 = __builtin_amdgcn_wmma_f32_16x16x32_f16(false, a1, false, b2, (short)0, acc12, false, false);
    acc03 = __builtin_amdgcn_wmma_f32_16x16x32_f16(false, a0, false, b3, (short)0, acc03, false, false);
    acc13 = __builtin_amdgcn_wmma_f32_16x16x32_f16(false, a1, false, b3, (short)0, acc13, false, false);
  }
  // C/D layout: VGPR rr -> row m0 + rr + 8*hs (+16 for second tile), col n0 + 16*j + r
  float* cp0 = C + (m0 + (size_t)hs * 8) * (size_t)Nc + n0 + r;
  float* cp1 = cp0 + (size_t)16 * Nc;
#pragma unroll
  for (int rr = 0; rr < 8; rr++) {
    cp0[(size_t)rr * Nc +  0] = acc00[rr];
    cp0[(size_t)rr * Nc + 16] = acc01[rr];
    cp0[(size_t)rr * Nc + 32] = acc02[rr];
    cp0[(size_t)rr * Nc + 48] = acc03[rr];
    cp1[(size_t)rr * Nc +  0] = acc10[rr];
    cp1[(size_t)rr * Nc + 16] = acc11[rr];
    cp1[(size_t)rr * Nc + 32] = acc12[rr];
    cp1[(size_t)rr * Nc + 48] = acc13[rr];
  }
}

// ---------------- utility / prep kernels ----------------
__global__ void zero_u32(uint32_t* p, size_t n) {
  size_t i = (size_t)blockIdx.x * 256 + threadIdx.x;
  size_t st = (size_t)gridDim.x * 256;
  for (; i < n; i += st) p[i] = 0u;
}

// transpose HxH f32 -> [c,k] f16
__global__ void pack_t(const float* __restrict__ W, _Float16* __restrict__ Wt) {
  int c = blockIdx.x, k = threadIdx.x;
  Wt[(size_t)c * HH + k] = (_Float16)W[(size_t)k * HH + c];
}
// transpose Hx4H f32, gate-interleaved: packed col 4h+g <- orig col g*H+h
__global__ void pack_tg(const float* __restrict__ W, _Float16* __restrict__ Wt) {
  int c = blockIdx.x, k = threadIdx.x;
  int h = c >> 2, g = c & 3;
  Wt[(size_t)c * HH + k] = (_Float16)W[(size_t)k * H4 + g * HH + h];
}
__global__ void pack_bg(const float* __restrict__ b, float* __restrict__ bp) {
  int c = threadIdx.x;  // 1024 threads
  bp[c] = b[(c & 3) * HH + (c >> 2)];
}

// embedding gather -> tok f16 [T, M1P, H] (pad rows pre-zeroed)
__global__ void tok_gather(const int* __restrict__ code, const float* __restrict__ embed,
                           _Float16* __restrict__ tok) {
  int idx = blockIdx.x;           // T*M1
  int t = idx / M1, m = idx % M1;
  int b = m / NS, s = m % NS;
  int tid = code[b * LL + s * TT + t];
  tok[((size_t)t * M1P + m) * HH + threadIdx.x] =
      (_Float16)embed[(size_t)tid * HH + threadIdx.x];
}

// stage-1 LSTM gating: z = z1 + z2 + b (gate-interleaved), update c,h
__global__ void lstm_gate1(const float* __restrict__ z1, const float* __restrict__ z2,
                           const float* __restrict__ bp, float* __restrict__ c,
                           float* __restrict__ h, _Float16* __restrict__ h16) {
  int m = blockIdx.x, hh = threadIdx.x;
  float4 a  = *(const float4*)(z1 + (size_t)m * H4 + 4 * hh);
  float4 bz = *(const float4*)(z2 + (size_t)m * H4 + 4 * hh);
  float4 bb = *(const float4*)(bp + 4 * hh);
  float zi = a.x + bz.x + bb.x, zf = a.y + bz.y + bb.y;
  float zg = a.z + bz.z + bb.z, zo = a.w + bz.w + bb.w;
  size_t idx = (size_t)m * HH + hh;
  float cc = c[idx];
  float c2 = sigf(zf) * cc + sigf(zi) * tanhf(zg);
  float h2 = sigf(zo) * tanhf(c2);
  c[idx] = c2; h[idx] = h2; h16[idx] = (_Float16)h2;
}

// prefix sums: pref[b,n,h] = sum_{m<n} stmt[b,m,h], n=0..96
__global__ void pref_k(const float* __restrict__ hs, float* __restrict__ pref) {
  int b = blockIdx.x, hh = threadIdx.x;
  float acc = 0.f;
  pref[((size_t)b * NN) * HH + hh] = 0.f;
  for (int n = 1; n < NN; n++) {
    acc += hs[((size_t)b * NS + (n - 1)) * HH + hh];
    pref[((size_t)(b * NN + n)) * HH + hh] = acc;
  }
}

// diff rows: (b,i,j) -> pref[j]-pref[i], f16
__global__ void diff_k(const float* __restrict__ pref, _Float16* __restrict__ dif) {
  int row = blockIdx.x;
  int b = row / (NN * NN), rem = row % (NN * NN);
  int i = rem / NN, j = rem % NN;
  int hh = threadIdx.x;
  float v = pref[((size_t)(b * NN + j)) * HH + hh] - pref[((size_t)(b * NN + i)) * HH + hh];
  dif[(size_t)row * HH + hh] = (_Float16)v;
}

// skip = C + b_se -> f16
__global__ void skipfin_k(const float* __restrict__ Cb, const float* __restrict__ bse,
                          _Float16* __restrict__ out) {
  size_t idx = (size_t)blockIdx.x * HH + threadIdx.x;
  out[idx] = (_Float16)(Cb[idx] + bse[threadIdx.x]);
}

// per-row reduction: out[row] = sum_h relu(C[row,h]+bias[h]) * wvec[h]
__global__ void rowdot_relu_k(const float* __restrict__ Cb, const float* __restrict__ bias,
                              const float* __restrict__ wvec, float* __restrict__ out) {
  __shared__ float sm[256];
  int row = blockIdx.x, hh = threadIdx.x;
  float v = Cb[(size_t)row * HH + hh] + bias[hh];
  v = fmaxf(v, 0.f) * wvec[hh];
  sm[hh] = v; __syncthreads();
  for (int s = 128; s > 0; s >>= 1) { if (hh < s) sm[hh] += sm[hh + s]; __syncthreads(); }
  if (hh == 0) out[row] = sm[0];
}

// masked softmax per (b,i): w[b,i,j] = p[b,i] * softmax_j(logits)
__global__ void logits_k(const float* __restrict__ Avec, const float* __restrict__ S2,
                         const float* __restrict__ bd1, const int* __restrict__ clen,
                         const float* __restrict__ pin, float* __restrict__ w, int layer) {
  __shared__ float sm[128];
  int bi = blockIdx.x;
  int b = bi / NN, i = bi % NN;
  int j = threadIdx.x;
  int len = clen[b] / TT;
  bool allowed = false;
  if (j < NN) {
    if (layer == 0)             allowed = (j == 1);
    else if (layer == NLAY - 1) allowed = (j == len);
    else                        allowed = ((j > i) && (i <= len)) || (j == len);
  }
  float lg = -INFINITY;
  if (allowed) lg = Avec[bi] + S2[(size_t)bi * NN + j] + bd1[0];
  sm[j] = lg; __syncthreads();
  for (int s = 64; s > 0; s >>= 1) { if (j < s) sm[j] = fmaxf(sm[j], sm[j + s]); __syncthreads(); }
  float mx = sm[0]; __syncthreads();
  float ex = allowed ? expf(lg - mx) : 0.f;
  sm[j] = ex; __syncthreads();
  for (int s = 64; s > 0; s >>= 1) { if (j < s) sm[j] += sm[j + s]; __syncthreads(); }
  float scale = pin[bi] / sm[0];
  if (j < NN) w[(size_t)bi * NN + j] = ex * scale;
}

// p_new[b,j] = sum_i w[b,i,j]; denom = p_new + 1e-7
__global__ void pden_k(const float* __restrict__ w, float* __restrict__ pout,
                       float* __restrict__ denom) {
  __shared__ float sm[128];
  int bj = blockIdx.x;
  int b = bj / NN, j = bj % NN;
  int i = threadIdx.x;
  float v = (i < NN) ? w[((size_t)(b * NN + i)) * NN + j] : 0.f;
  sm[i] = v; __syncthreads();
  for (int s = 64; s > 0; s >>= 1) { if (i < s) sm[i] += sm[i + s]; __syncthreads(); }
  if (i == 0) { pout[bj] = sm[0]; denom[bj] = sm[0] + 1e-7f; }
}

// gating + weighted aggregation over i for each (b,j,h)
__global__ void gate_k(const float* __restrict__ E0, const float* __restrict__ hWh,
                       const float* __restrict__ bep, const float* __restrict__ w,
                       const float* __restrict__ denom, const float* __restrict__ cin,
                       const float* __restrict__ hin, float* __restrict__ cout,
                       float* __restrict__ hout, _Float16* __restrict__ h16out) {
  int bj = blockIdx.x;
  int b = bj / NN, j = bj % NN;
  int hh = threadIdx.x;
  float4 bb = *(const float4*)(bep + 4 * hh);
  float csum = 0.f, hsum = 0.f;
  for (int i = 0; i < NN; i++) {
    size_t row = (size_t)(b * NN + i) * NN + j;
    float wv = w[row];
    if (wv == 0.f) continue;
    size_t mi = (size_t)(b * NN + i) * HH + hh;
    float ci = cin[mi], hi = hin[mi];
    float cp, hp;
    if (j > i) {
      float4 ze = *(const float4*)(E0 + row * H4 + 4 * hh);
      float4 zw = *(const float4*)(hWh + (size_t)(b * NN + i) * H4 + 4 * hh);
      float zi = ze.x + zw.x + bb.x, zf = ze.y + zw.y + bb.y;
      float zg = ze.z + zw.z + bb.z, zo = ze.w + zw.w + bb.w;
      cp = sigf(zf) * ci + sigf(zi) * tanhf(zg);
      hp = sigf(zo) * tanhf(cp);
    } else { cp = ci; hp = hi; }
    csum += wv * cp; hsum += wv * hp;
  }
  float dn = denom[bj];
  size_t mo = (size_t)bj * HH + hh;
  float cv = csum / dn, hv = hsum / dn;
  cout[mo] = cv; hout[mo] = hv; h16out[mo] = (_Float16)hv;
}

__global__ void initch_k(const float* __restrict__ ia, const float* __restrict__ ib,
                         float* __restrict__ c, float* __restrict__ h,
                         _Float16* __restrict__ h16) {
  size_t idx = (size_t)blockIdx.x * HH + threadIdx.x;
  float av = ia[threadIdx.x], bv = ib[threadIdx.x];
  c[idx] = av; h[idx] = bv; h16[idx] = (_Float16)bv;
}

__global__ void setp_k(float* __restrict__ p) {
  int t = blockIdx.x * 256 + threadIdx.x;
  if (t < MH) p[t] = ((t % NN) == 0) ? 1.0f : 0.0f;
}

__global__ void copyout_k(const float* __restrict__ h, float* __restrict__ out) {
  size_t idx = (size_t)blockIdx.x * HH + threadIdx.x;
  out[idx] = h[idx];
}

// ---------------- host side ----------------
extern "C" void kernel_launch(void* const* d_in, const int* in_sizes, int n_in,
                              void* d_out, int out_size, void* d_ws, size_t ws_size,
                              hipStream_t stream) {
  (void)in_sizes; (void)n_in; (void)out_size; (void)ws_size;
  const int*   code  = (const int*)d_in[0];
  const int*   clen  = (const int*)d_in[1];
  const float* embed = (const float*)d_in[2];
  const float* Wx_s  = (const float*)d_in[3];
  const float* Wh_s  = (const float*)d_in[4];
  const float* b_s   = (const float*)d_in[5];
  const float* W_se  = (const float*)d_in[6];
  const float* b_se  = (const float*)d_in[7];
  const float* Wx_e  = (const float*)d_in[8];
  const float* Wh_e  = (const float*)d_in[9];
  const float* b_e   = (const float*)d_in[10];
  const float* W_hk  = (const float*)d_in[11];
  const float* b_hk  = (const float*)d_in[12];
  const float* W_sd  = (const float*)d_in[13];
  const float* b_sd  = (const float*)d_in[14];
  const float* W_d1  = (const float*)d_in[15];
  const float* b_d1  = (const float*)d_in[16];
  const float* inita = (const float*)d_in[17];
  const float* initb = (const float*)d_in[18];

  char* base = (char*)d_ws;
  size_t off = 0;
  auto alloc = [&](size_t bytes) -> char* {
    char* p = base + off;
    off = (off + bytes + 255) & ~(size_t)255;
    return p;
  };

  float*    E0    = (float*)   alloc((size_t)MSP * H4 * 4);   // also scratch for skip/keys GEMM C
  _Float16* sk16  = (_Float16*)alloc((size_t)MSP * HH * 2);
  _Float16* df16  = (_Float16*)alloc((size_t)MSP * HH * 2);
  _Float16* tokx  = (_Float16*)alloc((size_t)TT * M1P * HH * 2);
  float*    zb1   = (float*)   alloc((size_t)M1P * H4 * 4);
  float*    zb2   = (float*)   alloc((size_t)M1P * H4 * 4);
  float*    hs1   = (float*)   alloc((size_t)M1 * HH * 4);
  float*    cs1   = (float*)   alloc((size_t)M1 * HH * 4);
  _Float16* hs16  = (_Float16*)alloc((size_t)M1P * HH * 2);
  float*    pref  = (float*)   alloc((size_t)MH * HH * 4);
  float*    cA    = (float*)   alloc((size_t)MHP * HH * 4);
  float*    cBf   = (float*)   alloc((size_t)MHP * HH * 4);
  float*    hA    = (float*)   alloc((size_t)MHP * HH * 4);
  float*    hB    = (float*)   alloc((size_t)MHP * HH * 4);
  _Float16* h16A  = (_Float16*)alloc((size_t)MHP * HH * 2);
  _Float16* h16B  = (_Float16*)alloc((size_t)MHP * HH * 2);
  float*    hWh   = (float*)   alloc((size_t)MHP * H4 * 4);
  float*    hkey  = (float*)   alloc((size_t)MHP * HH * 4);
  float*    Avec  = (float*)   alloc((size_t)MHP * 4);
  float*    S2    = (float*)   alloc((size_t)MS * 4);
  float*    wbuf  = (float*)   alloc((size_t)MS * 4);
  float*    pA    = (float*)   alloc((size_t)MH * 4);
  float*    pB    = (float*)   alloc((size_t)MH * 4);
  float*    denom = (float*)   alloc((size_t)MH * 4);
  _Float16* WxSt  = (_Float16*)alloc((size_t)H4 * HH * 2);
  _Float16* WhSt  = (_Float16*)alloc((size_t)H4 * HH * 2);
  _Float16* WSEt  = (_Float16*)alloc((size_t)HH * HH * 2);
  _Float16* WSDt  = (_Float16*)alloc((size_t)HH * HH * 2);
  _Float16* WxEt  = (_Float16*)alloc((size_t)H4 * HH * 2);
  _Float16* WhEt  = (_Float16*)alloc((size_t)H4 * HH * 2);
  _Float16* WHKt  = (_Float16*)alloc((size_t)HH * HH * 2);
  float*    bsP   = (float*)   alloc(H4 * 4);
  float*    beP   = (float*)   alloc(H4 * 4);

  auto zero = [&](void* p, size_t bytes) {
    size_t n = bytes / 4;
    int blocks = (int)(((n + 255) / 256) > 4096 ? 4096 : ((n + 255) / 256));
    zero_u32<<<blocks, 256, 0, stream>>>((uint32_t*)p, n);
  };
  auto gemm = [&](const _Float16* A, const _Float16* Bt, float* C, int Mp, int Nc) {
    dim3 g(Nc / 64, Mp / 256);
    gemm_wmma<<<g, 256, 0, stream>>>(A, Bt, C, Nc);
  };

  // ---- init / zero padded buffers ----
  zero(df16, (size_t)MSP * HH * 2);
  zero(sk16, (size_t)MSP * HH * 2);
  zero(tokx, (size_t)TT * M1P * HH * 2);
  zero(hs1,  (size_t)M1 * HH * 4);
  zero(cs1,  (size_t)M1 * HH * 4);
  zero(hs16, (size_t)M1P * HH * 2);
  zero(h16A, (size_t)MHP * HH * 2);
  zero(h16B, (size_t)MHP * HH * 2);

  // ---- pack weights (transposed, gate-interleaved for 4H) ----
  pack_tg<<<H4, HH, 0, stream>>>(Wx_s, WxSt);
  pack_tg<<<H4, HH, 0, stream>>>(Wh_s, WhSt);
  pack_t <<<HH, HH, 0, stream>>>(W_se, WSEt);
  pack_t <<<HH, HH, 0, stream>>>(W_sd, WSDt);
  pack_tg<<<H4, HH, 0, stream>>>(Wx_e, WxEt);
  pack_tg<<<H4, HH, 0, stream>>>(Wh_e, WhEt);
  pack_t <<<HH, HH, 0, stream>>>(W_hk, WHKt);
  pack_bg<<<1, H4, 0, stream>>>(b_s, bsP);
  pack_bg<<<1, H4, 0, stream>>>(b_e, beP);

  // ---- stage 1: statement LSTM over T tokens ----
  tok_gather<<<TT * M1, HH, 0, stream>>>(code, embed, tokx);
  for (int t = 0; t < TT; t++) {
    gemm(tokx + (size_t)t * M1P * HH, WxSt, zb1, M1P, H4);
    gemm(hs16, WhSt, zb2, M1P, H4);
    lstm_gate1<<<M1, HH, 0, stream>>>(zb1, zb2, bsP, cs1, hs1, hs16);
  }

  // ---- skip encoder (loop-invariant) ----
  pref_k<<<BB, HH, 0, stream>>>(hs1, pref);
  diff_k<<<MS, HH, 0, stream>>>(pref, df16);
  gemm(df16, WSEt, E0, MSP, HH);                         // diff @ W_se -> E0 scratch
  skipfin_k<<<MS, HH, 0, stream>>>(E0, b_se, sk16);      // + b_se -> skip f16

  gemm(sk16, WSDt, E0, MSP, HH);                         // skip @ W_sd -> E0 scratch
  rowdot_relu_k<<<MS, HH, 0, stream>>>(E0, b_sd, W_d1 + HH, S2);  // S2 (loop-invariant)

  gemm(sk16, WxEt, E0, MSP, H4);                         // E0 = skip @ Wx_e (gate-interleaved)

  // ---- execution loop ----
  initch_k<<<MH, HH, 0, stream>>>(inita, initb, cA, hA, h16A);
  setp_k<<<2, 256, 0, stream>>>(pA);

  for (int layer = 0; layer < NLAY; layer++) {
    const bool even = (layer % 2) == 0;
    const float*    cin   = even ? cA  : cBf;
    const float*    hin   = even ? hA  : hB;
    const _Float16* h16in = even ? h16A : h16B;
    float*          coutp = even ? cBf : cA;
    float*          houtp = even ? hB  : hA;
    _Float16*       h16o  = even ? h16B : h16A;
    const float*    pin   = even ? pA : pB;
    float*          pout  = even ? pB : pA;

    gemm(h16in, WhEt, hWh, MHP, H4);                     // h @ Wh_e (gate-interleaved)
    gemm(h16in, WHKt, hkey, MHP, HH);                    // h @ W_hk
    rowdot_relu_k<<<MH, HH, 0, stream>>>(hkey, b_hk, W_d1, Avec);
    logits_k<<<MH, 128, 0, stream>>>(Avec, S2, b_d1, clen, pin, wbuf, layer);
    pden_k<<<MH, 128, 0, stream>>>(wbuf, pout, denom);
    gate_k<<<MH, HH, 0, stream>>>(E0, hWh, beP, wbuf, denom, cin, hin,
                                  coutp, houtp, h16o);
  }

  // NLAY = 6 (even) -> final h is in hA
  copyout_k<<<MH, HH, 0, stream>>>(hA, (float*)d_out);
}
